// CausalSelfAttention_28973849378864
// MI455X (gfx1250) — compile-verified, matrix operand order checked by reading, not run on hardware
//
#include <hip/hip_runtime.h>
#include <hip/hip_bf16.h>
#include <math.h>

// Problem constants (from reference)
#define DIMN 2048
#define NH   16
#define NKV  4
#define HD   128
#define BB   4
#define SS   2048

typedef __attribute__((ext_vector_type(16))) __bf16 bf16x16;
typedef __attribute__((ext_vector_type(8)))  __bf16 bf16x8;
typedef __attribute__((ext_vector_type(8)))  float  f32x8;
typedef __attribute__((ext_vector_type(8)))  unsigned short u16x8;

__device__ __forceinline__ unsigned short f2bf(float f) {
  union { float f; unsigned u; } v; v.f = f;
  unsigned u = v.u;
  u += 0x7FFFu + ((u >> 16) & 1u);   // round to nearest even
  return (unsigned short)(u >> 16);
}
__device__ __forceinline__ float bf2f(unsigned short h) {
  union { unsigned u; float f; } v; v.u = ((unsigned)h) << 16;
  return v.f;
}

union FragBF { bf16x16 v; bf16x8 h2[2]; unsigned w[8]; };

// Load one bf16 WMMA fragment: per-lane two 16B chunks at K-offsets
// {half*8} and {16+half*8} (ISA 16-bit matrix layout, used for A and B).
__device__ __forceinline__ FragBF load_frag(const unsigned short* p, int off) {
  FragBF f;
  f.h2[0] = *(const bf16x8*)(p + off);
  f.h2[1] = *(const bf16x8*)(p + off + 16);
  return f;
}

// ---------------- CDNA5 async global->LDS copy (ASYNCcnt-tracked) ----------
// GV mode: per-lane 16B  copy from global address (vaddr) to LDS offset (vdst).
// Generic->LDS address is defined by the ISA as truncation to addr[31:0].
__device__ __forceinline__ void async_cp16(unsigned lds_off,
                                           const unsigned short* g) {
  asm volatile("global_load_async_to_lds_b128 %0, %1, off"
               :: "v"(lds_off), "v"((const void*)g) : "memory");
}
__device__ __forceinline__ void async_wait0() {
  asm volatile("s_wait_asynccnt 0x0" ::: "memory");
}
__device__ __forceinline__ unsigned lds_off32(const void* p) {
  return (unsigned)(uintptr_t)p;
}

// ---------------------------------------------------------------- convert
__global__ void k_f32_to_bf16(const float* __restrict__ in,
                              unsigned short* __restrict__ out, int n) {
  int i = blockIdx.x * blockDim.x + threadIdx.x;
  if (i < n) out[i] = f2bf(in[i]);
}

// ---------------------------------------------------------------- GEMM
// C[M,N] = A[M,K] * B[N,K]^T   (A,B bf16 row-major; C fp32 or bf16)
// 256 threads, 128x128 block tile, 8 waves as 2x4 (64x32/wave), K-step 32,
// double-buffered LDS fed by async global->LDS copies.
template<bool F32OUT>
__global__ __launch_bounds__(256) void k_gemm_bt(
    const unsigned short* __restrict__ A, const unsigned short* __restrict__ Bm,
    float* __restrict__ Cf, unsigned short* __restrict__ Cb,
    int M, int N, int K) {
  __shared__ unsigned short sa[2][128 * 40];   // 32 K elems + pad, x2 buffers
  __shared__ unsigned short sb[2][128 * 40];
  const int tid  = threadIdx.x;
  const int lane = tid & 31, wave = tid >> 5;
  const int hf   = lane >> 4, ln = lane & 15;
  const int wm   = wave >> 2, wn = wave & 3;
  const int m0 = blockIdx.y * 128, n0 = blockIdx.x * 128;

  const f32x8 zero = {0.f,0.f,0.f,0.f,0.f,0.f,0.f,0.f};
  f32x8 acc[4][2];
  for (int t = 0; t < 4; ++t)
    for (int u = 0; u < 2; ++u) acc[t][u] = zero;

  const int row = tid >> 1, seg = (tid & 1) * 16;
  const unsigned short* gA = A  + (size_t)(m0 + row) * K + seg;
  const unsigned short* gB = Bm + (size_t)(n0 + row) * K + seg;

  auto stage = [&](int bufi, int k0) {
    unsigned la = lds_off32(&sa[bufi][row * 40 + seg]);
    unsigned lb = lds_off32(&sb[bufi][row * 40 + seg]);
    async_cp16(la,      gA + k0);
    async_cp16(la + 16, gA + k0 + 8);
    async_cp16(lb,      gB + k0);
    async_cp16(lb + 16, gB + k0 + 8);
  };

  const int NIT = K / 32;
  stage(0, 0);
  for (int it = 0; it < NIT; ++it) {
    async_wait0();
    __syncthreads();                       // tiles for buf[it&1] visible
    if (it + 1 < NIT) stage((it + 1) & 1, (it + 1) * 32);

    const unsigned short* pa = sa[it & 1];
    const unsigned short* pb = sb[it & 1];
    FragBF bfr[2], afr[4];
    for (int u = 0; u < 2; ++u)
      bfr[u] = load_frag(pb, (wn * 32 + u * 16 + ln) * 40 + hf * 8);
    for (int t = 0; t < 4; ++t)
      afr[t] = load_frag(pa, (wm * 64 + t * 16 + ln) * 40 + hf * 8);
    for (int t = 0; t < 4; ++t)
      for (int u = 0; u < 2; ++u)
        acc[t][u] = __builtin_amdgcn_wmma_f32_16x16x32_bf16(
            false, afr[t].v, false, bfr[u].v, (short)0, acc[t][u], false, false);
    __syncthreads();                       // protect buf[it&1] before restage
  }

  for (int t = 0; t < 4; ++t)
    for (int u = 0; u < 2; ++u) {
      int gc = n0 + wn * 32 + u * 16 + ln;
      for (int i = 0; i < 8; ++i) {
        int gr = m0 + wm * 64 + t * 16 + hf * 8 + i;
        if constexpr (F32OUT) Cf[(size_t)gr * N + gc] = acc[t][u][i];
        else                  Cb[(size_t)gr * N + gc] = f2bf(acc[t][u][i]);
      }
    }
}

// ---------------------------------------------------------------- RoPE
__global__ void k_rope_q(const unsigned short* __restrict__ qp,  // (B*S, NH*HD)
                         unsigned short* __restrict__ qt) {      // (B,NH,S,HD)
  int idx = blockIdx.x * blockDim.x + threadIdx.x;
  const int total = BB * SS * NH * (HD / 2);
  if (idx >= total) return;
  int i   = idx & 63;
  int hh  = (idx >> 6) & 15;
  int row = idx >> 10;
  int pos = row & (SS - 1);
  int b   = row >> 11;
  float freq = __expf(-(float)i * (9.210340371976184f / 64.0f)); // 10000^(-i/64)
  float ang = (float)pos * freq;
  float sn, cs; __sincosf(ang, &sn, &cs);
  const unsigned short* src = qp + (size_t)row * DIMN + hh * HD;
  float lo = bf2f(src[i]), hi = bf2f(src[i + 64]);
  unsigned short* dst = qt + (((size_t)(b * NH + hh) * SS) + pos) * HD;
  dst[i]      = f2bf(lo * cs - hi * sn);
  dst[i + 64] = f2bf(hi * cs + lo * sn);
}

__global__ void k_rope_k(const unsigned short* __restrict__ kvp, // (B*S, 1024)
                         unsigned short* __restrict__ kt) {      // (B,NKV,S,HD)
  int idx = blockIdx.x * blockDim.x + threadIdx.x;
  const int total = BB * SS * NKV * (HD / 2);
  if (idx >= total) return;
  int i   = idx & 63;
  int kv  = (idx >> 6) & 3;
  int row = idx >> 8;
  int pos = row & (SS - 1);
  int b   = row >> 11;
  float freq = __expf(-(float)i * (9.210340371976184f / 64.0f));
  float ang = (float)pos * freq;
  float sn, cs; __sincosf(ang, &sn, &cs);
  const unsigned short* src = kvp + (size_t)row * (2 * NKV * HD) + kv * HD;
  float lo = bf2f(src[i]), hi = bf2f(src[i + 64]);
  unsigned short* dst = kt + (((size_t)(b * NKV + kv) * SS) + pos) * HD;
  dst[i]      = f2bf(lo * cs - hi * sn);
  dst[i + 64] = f2bf(hi * cs + lo * sn);
}

__global__ void k_v_transpose(const unsigned short* __restrict__ kvp, // (B*S,1024)
                              unsigned short* __restrict__ vt) {      // (B,NKV,HD,S)
  int idx = blockIdx.x * blockDim.x + threadIdx.x;
  const int total = BB * SS * NKV * HD;
  if (idx >= total) return;
  int d   = idx & 127;
  int kv  = (idx >> 7) & 3;
  int row = idx >> 9;
  int pos = row & (SS - 1);
  int b   = row >> 11;
  vt[(((size_t)(b * NKV + kv) * HD) + d) * SS + pos] =
      kvp[(size_t)row * (2 * NKV * HD) + NKV * HD + kv * HD + d];
}

// ---------------------------------------------------------------- attention
// Flash-attention. 128 threads = 4 waves; wave owns 16 q rows (64/WG).
// Scores computed transposed: T = K_tile(16x128) * Q^T, so the C-layout of T
// (lane = q, VGPRs = keys) is exactly the A-fragment layout needed for P*V.
// K/V tiles double-buffered in LDS, streamed with async global->LDS copies.
__global__ __launch_bounds__(128) void k_attn(
    const unsigned short* __restrict__ qt,   // (B,NH,S,HD)
    const unsigned short* __restrict__ kt,   // (B,NKV,S,HD)
    const unsigned short* __restrict__ vt,   // (B,NKV,HD,S)
    unsigned short* __restrict__ o) {        // (B,S,NH,HD)
  __shared__ unsigned short lk[2][32 * 136]; // K tile: 32 keys x 128 d (+pad)
  __shared__ unsigned short lv[2][128 * 40]; // V tile: 128 d x 32 keys (+pad)
  const int tid  = threadIdx.x;
  const int lane = tid & 31, wave = tid >> 5;
  const int hf   = lane >> 4, ln = lane & 15;
  const int q0 = blockIdx.x * 64;
  const int h  = blockIdx.y;
  const int b  = blockIdx.z;
  const int kv = h >> 2;
  const int qw = q0 + wave * 16;   // wave's q base
  const int qrow = qw + ln;        // q column this lane owns in T layout

  // Q fragments (B operand of T = K * Q^T): 4 chunks of K(d)=32
  FragBF qf[4];
  const unsigned short* qbase = qt + (((size_t)(b * NH + h) * SS) + qrow) * HD;
  for (int c = 0; c < 4; ++c) {
    qf[c].h2[0] = *(const bf16x8*)(qbase + c * 32 + hf * 8);
    qf[c].h2[1] = *(const bf16x8*)(qbase + c * 32 + hf * 8 + 16);
  }

  const f32x8 zero = {0.f,0.f,0.f,0.f,0.f,0.f,0.f,0.f};
  f32x8 oacc[8];
  for (int dt = 0; dt < 8; ++dt) oacc[dt] = zero;
  float mstat = -3.0e38f, lstat = 0.0f;
  const float sc = 0.08838834764831845f;  // 1/sqrt(128)

  const unsigned short* kbase = kt + (size_t)(b * NKV + kv) * SS * HD;
  const unsigned short* vbase = vt + (size_t)(b * NKV + kv) * HD * SS;

  const int kr = tid >> 2, seg = (tid & 3) * 32;  // K-tile staging map
  auto stage = [&](int bufi, int j) {
    const unsigned short* g = kbase + (size_t)(j + kr) * HD + seg;
    unsigned lkk = lds_off32(&lk[bufi][kr * 136 + seg]);
    async_cp16(lkk,      g);
    async_cp16(lkk + 16, g + 8);
    async_cp16(lkk + 32, g + 16);
    async_cp16(lkk + 48, g + 24);
    const unsigned short* gv = vbase + (size_t)tid * SS + j;
    unsigned lvv = lds_off32(&lv[bufi][tid * 40]);
    async_cp16(lvv,      gv);
    async_cp16(lvv + 16, gv + 8);
    async_cp16(lvv + 32, gv + 16);
    async_cp16(lvv + 48, gv + 24);
  };

  const int nit = (q0 + 64) / 32;   // key tiles: 0 .. q0+63
  stage(0, 0);
  for (int it = 0; it < nit; ++it) {
    const int j = it * 32;
    async_wait0();
    __syncthreads();
    if (it + 1 < nit) stage((it + 1) & 1, j + 32);

    if (j <= qw + 15) {   // wave still has unmasked keys in this tile
      const unsigned short* pk = lk[it & 1];
      const unsigned short* pv = lv[it & 1];

      // preload all 8 K fragments, then burst the 8 QK^T WMMAs
      FragBF kf[8];
      for (int c = 0; c < 4; ++c) {
        kf[c]     = load_frag(pk, ln * 136 + c * 32 + hf * 8);
        kf[4 + c] = load_frag(pk, (16 + ln) * 136 + c * 32 + hf * 8);
      }
      f32x8 t0 = zero, t1 = zero;   // T for keys j..j+15 and j+16..j+31
      for (int c = 0; c < 4; ++c) {
        t0 = __builtin_amdgcn_wmma_f32_16x16x32_bf16(
            false, kf[c].v, false, qf[c].v, (short)0, t0, false, false);
        t1 = __builtin_amdgcn_wmma_f32_16x16x32_bf16(
            false, kf[4 + c].v, false, qf[c].v, (short)0, t1, false, false);
      }
      // scale + causal mask (key > q  ->  -inf surrogate)
      float p0[8], p1[8];
      float rowm = -3.0e38f;
      for (int i = 0; i < 8; ++i) {
        int k0i = j + hf * 8 + i;
        int k1i = j + 16 + hf * 8 + i;
        p0[i] = (k0i <= qrow) ? t0[i] * sc : -3.0e38f;
        p1[i] = (k1i <= qrow) ? t1[i] * sc : -3.0e38f;
        rowm = fmaxf(rowm, fmaxf(p0[i], p1[i]));
      }
      rowm = fmaxf(rowm, __shfl_xor(rowm, 16, 32));
      float mnew  = fmaxf(mstat, rowm);
      float alpha = __expf(mstat - mnew);
      float rsum = 0.0f;
      for (int i = 0; i < 8; ++i) {
        p0[i] = __expf(p0[i] - mnew);
        p1[i] = __expf(p1[i] - mnew);
        rsum += p0[i] + p1[i];
      }
      rsum += __shfl_xor(rsum, 16, 32);
      lstat = lstat * alpha + rsum;
      mstat = mnew;

      // pack P into an A-fragment (keys already sit in the right lanes)
      FragBF pf;
      for (int i = 0; i < 4; ++i) {
        pf.w[i]     = ((unsigned)f2bf(p0[2 * i + 1]) << 16) | f2bf(p0[2 * i]);
        pf.w[4 + i] = ((unsigned)f2bf(p1[2 * i + 1]) << 16) | f2bf(p1[2 * i]);
      }
      // broadcast per-q rescale factors into accumulator-row order
      float al8[8];
      for (int i = 0; i < 8; ++i) al8[i] = __shfl(alpha, hf * 8 + i, 32);

      // P*V, software-pipelined: fetch V fragment dt+1 under WMMA dt
      FragBF vf = load_frag(pv, (0 * 16 + ln) * 40 + hf * 8);
      for (int dt = 0; dt < 8; ++dt) {
        const int dn = (dt < 7) ? dt + 1 : 7;
        FragBF vnext = load_frag(pv, (dn * 16 + ln) * 40 + hf * 8);
        f32x8 a = oacc[dt];
        for (int i = 0; i < 8; ++i) a[i] *= al8[i];
        oacc[dt] = __builtin_amdgcn_wmma_f32_16x16x32_bf16(
            false, pf.v, false, vf.v, (short)0, a, false, false);
        vf = vnext;
      }
    }
    __syncthreads();
  }

  float linv[8];
  for (int i = 0; i < 8; ++i) {
    float lsum = __shfl(lstat, hf * 8 + i, 32);
    linv[i] = 1.0f / lsum;
  }
  for (int dt = 0; dt < 8; ++dt)
    for (int i = 0; i < 8; ++i) {
      int q = qw + hf * 8 + i;
      int d = dt * 16 + ln;
      o[(((size_t)(b * SS + q) * NH) + h) * HD + d] = f2bf(oacc[dt][i] * linv[i]);
    }
}

// ---------------------------------------------------------------- launcher
extern "C" void kernel_launch(void* const* d_in, const int* in_sizes, int n_in,
                              void* d_out, int out_size, void* d_ws, size_t ws_size,
                              hipStream_t stream) {
  (void)in_sizes; (void)n_in; (void)out_size; (void)ws_size;
  const float* x   = (const float*)d_in[0];
  const float* Wq  = (const float*)d_in[1];
  const float* Wkv = (const float*)d_in[2];
  const float* Wo  = (const float*)d_in[3];

  char* ws = (char*)d_ws;
  unsigned short* xb     = (unsigned short*)(ws + 0);           // 32 MB
  unsigned short* wqb    = (unsigned short*)(ws + 33554432);    //  8 MB
  unsigned short* wkvb   = (unsigned short*)(ws + 41943040);    //  4 MB
  unsigned short* wob    = (unsigned short*)(ws + 46137344);    //  8 MB
  unsigned short* qproj  = (unsigned short*)(ws + 54525952);    // 32 MB
  unsigned short* kvproj = (unsigned short*)(ws + 88080384);    // 16 MB
  unsigned short* qtb    = (unsigned short*)(ws + 104857600);   // 32 MB
  unsigned short* ktb    = (unsigned short*)(ws + 138412032);   //  8 MB
  unsigned short* vtb    = (unsigned short*)(ws + 146800640);   //  8 MB
  unsigned short* attn   = (unsigned short*)(ws + 155189248);   // 32 MB

  const int nx   = BB * SS * DIMN;
  const int nwq  = DIMN * DIMN;
  const int nwkv = 2 * NKV * HD * DIMN;
  k_f32_to_bf16<<<(nx   + 255) / 256, 256, 0, stream>>>(x,   xb,   nx);
  k_f32_to_bf16<<<(nwq  + 255) / 256, 256, 0, stream>>>(Wq,  wqb,  nwq);
  k_f32_to_bf16<<<(nwkv + 255) / 256, 256, 0, stream>>>(Wkv, wkvb, nwkv);
  k_f32_to_bf16<<<(nwq  + 255) / 256, 256, 0, stream>>>(Wo,  wob,  nwq);

  dim3 blk(256);
  dim3 gq(DIMN / 128, (BB * SS) / 128);
  k_gemm_bt<false><<<gq, blk, 0, stream>>>(xb, wqb, nullptr, qproj,
                                           BB * SS, DIMN, DIMN);
  dim3 gkv((2 * NKV * HD) / 128, (BB * SS) / 128);
  k_gemm_bt<false><<<gkv, blk, 0, stream>>>(xb, wkvb, nullptr, kvproj,
                                            BB * SS, 2 * NKV * HD, DIMN);

  const int nq = BB * SS * NH  * (HD / 2);
  const int nk = BB * SS * NKV * (HD / 2);
  const int nv = BB * SS * NKV * HD;
  k_rope_q<<<(nq + 255) / 256, 256, 0, stream>>>(qproj,  qtb);
  k_rope_k<<<(nk + 255) / 256, 256, 0, stream>>>(kvproj, ktb);
  k_v_transpose<<<(nv + 255) / 256, 256, 0, stream>>>(kvproj, vtb);

  dim3 ga(SS / 64, NH, BB);
  k_attn<<<ga, dim3(128), 0, stream>>>(qtb, ktb, vtb, attn);

  k_gemm_bt<true><<<gq, blk, 0, stream>>>(attn, wob, (float*)d_out, nullptr,
                                          BB * SS, DIMN, DIMN);
}
